// TRANS_MIL_38938173505836
// MI455X (gfx1250) — matrix operand order repaired and checked by reading, not compiled, not run on hardware
//
#include <hip/hip_runtime.h>

typedef __bf16 bf16_t;
typedef __attribute__((ext_vector_type(16))) __bf16 v16bf;
typedef __attribute__((ext_vector_type(8)))  float  v8f;

static constexpr int NPATCH = 40000;
static constexpr int INDIM  = 1024;
static constexpr int DIMC   = 512;
static constexpr int PADR   = 191;        // (256 - 40001 % 256) % 256
static constexpr int NPAD   = 40192;      // 40001 + 191
static constexpr int NTOK   = 40001;
static constexpr int LVAL   = 157;        // NPAD / 256
static constexpr int MLM    = 256;
static constexpr int NHEAD  = 8;
static constexpr int DHEAD  = 64;

static constexpr int BMt = 128, BNt = 64, BKt = 32;

// ---------------- WMMA helpers (contiguous-k layouts, wave32) -------------------
// Fragments are staged into LDS in "fragment-ready" order: lane L's 16 bf16
// elements are contiguous (32 bytes), so a fragment load is one 32B vector
// load (2 x ds_load_b128).
//
// A(16x32) element (r,k): lane = r + 16*(k>=16),     elem = k&15
// B(32x16) element (k,n): lane = n + 16*(k>=16),     elem = k&15
// Note: 8 consecutive K values (aligned) never cross a swizzle boundary, so
// A tiles can be staged as aligned 16-byte chunks on both sides.

static __device__ inline v16bf frag_ld(const bf16_t* base) {
  return *(const v16bf*)(base + ((threadIdx.x & 31) << 4));
}

static __device__ inline v8f wmma_bf16(v16bf a, v16bf b, v8f c) {
  return __builtin_amdgcn_wmma_f32_16x16x32_bf16(false, a, false, b, (short)0, c, false, false);
}

// ---------------- Generic batched bf16 WMMA GEMM ------------------------------
// C[M,N] = scale * (A[M,K] @ B[K,N]) (+bias) (+relu) ; optional accumulate into
// Cf, optional bf16 store Cb. Per-batch strides sA/sB/sC (elements).
// All A/B base pointers and leading dims are 8-element (16B) aligned.

__global__ __launch_bounds__(256)
void k_gemm(const bf16_t* __restrict__ A, int lda, long long sA,
            const bf16_t* __restrict__ B, int ldb, long long sB,
            float* Cf, bf16_t* Cb, int ldc, long long sC,
            int M, int N, int K,
            const float* __restrict__ bias, int relu, int accum, float scale)
{
  // fragment-ready LDS: As[wid][lane][e], Bs[ct][lane][e]
  __shared__ __align__(32) bf16_t As[BMt * BKt];   // 8 subtiles * 512
  __shared__ __align__(32) bf16_t Bs[BKt * BNt];   // 4 subtiles * 512
  int bz = blockIdx.z;
  const bf16_t* Ab = A + (long long)bz * sA;
  const bf16_t* Bb = B + (long long)bz * sB;
  long long coff = (long long)bz * sC;
  int m0 = blockIdx.y * BMt;
  int n0 = blockIdx.x * BNt;
  int tid = threadIdx.x;
  int wid = tid >> 5;

  v8f acc[4];
#pragma unroll
  for (int t = 0; t < 4; ++t)
#pragma unroll
    for (int e = 0; e < 8; ++e) acc[t][e] = 0.f;

  for (int k0 = 0; k0 < K; k0 += BKt) {
    // prefetch next K tile into caches (global_prefetch_b8)
    {
      int kn = k0 + BKt;
      if (kn < K) {
        long long pr = m0 + (tid >> 1); if (pr >= M) pr = M - 1;
        __builtin_prefetch(Ab + pr * lda + kn, 0, 1);
        int pk = kn + (tid & 31); if (pk >= K) pk = K - 1;
        __builtin_prefetch(Bb + (long long)pk * ldb + n0, 0, 1);
      }
    }
    bool interior = (m0 + BMt <= M) && (n0 + BNt <= N) && (k0 + BKt <= K);
    if (interior) {
      // A: 512 aligned 16B chunks (8 bf16), 2 per thread, b128 both sides
#pragma unroll
      for (int it = 0; it < 2; ++it) {
        int g = tid + it * 256;          // chunk index 0..511
        int rg = g >> 2;                 // 0..127
        int kko = (g & 3) << 3;          // 0,8,16,24
        const bf16_t* src = Ab + (long long)(m0 + rg) * lda + (k0 + kko);
        int w2 = rg >> 4, r = rg & 15;
        int lane = r + 16 * (kko >> 4);
        int e = kko & 15;
        *(uint4*)&As[(w2 << 9) + (lane << 4) + e] = *(const uint4*)src;
      }
      // B: 1024 b32 loads (pairs along N) + swizzled b16 stores, 4 per thread
#pragma unroll
      for (int it = 0; it < 4; ++it) {
        int g = tid + it * 256;          // pair index
        int kk = g >> 5;                 // 0..31
        int np = (g & 31) << 1;          // 0,2..62
        const bf16_t* src = Bb + (long long)(k0 + kk) * ldb + (n0 + np);
        bf16_t v0 = src[0], v1 = src[1];
        int ct = np >> 4, nn = np & 15;
        int base = (ct << 9) + (kk & 15) + ((16 * (kk >> 4)) << 4);
        Bs[base + (nn << 4)] = v0;
        Bs[base + ((nn + 1) << 4)] = v1;
      }
    } else {
      // guarded edge path (rare: last M tile only)
#pragma unroll
      for (int it = 0; it < (BMt * BKt) / 512; ++it) {
        int g = tid + it * 256;          // pair index
        int rg = g >> 4;                 // 0..127
        int kkp = (g & 15) << 1;         // 0,2..30
        bf16_t v0 = (bf16_t)0.f, v1 = (bf16_t)0.f;
        long long row = m0 + rg;
        if (row < M) {
          if (k0 + kkp     < K) v0 = Ab[row * lda + (k0 + kkp)];
          if (k0 + kkp + 1 < K) v1 = Ab[row * lda + (k0 + kkp + 1)];
        }
        int w2 = rg >> 4, r = rg & 15;
        int lane = r + 16 * (kkp >> 4);
        int e = kkp & 15;
        bf16_t* d = &As[(w2 << 9) + (lane << 4) + e];
        d[0] = v0; d[1] = v1;
      }
#pragma unroll
      for (int it = 0; it < (BKt * BNt) / 256; ++it) {
        int g = tid + it * 256;
        int kk = g >> 6, n6 = g & 63;
        bf16_t v = (bf16_t)0.f;
        if (k0 + kk < K && n0 + n6 < N) v = Bb[(long long)(k0 + kk) * ldb + (n0 + n6)];
        int ct = n6 >> 4, nn = n6 & 15;
        Bs[(ct << 9) + ((nn + 16 * (kk >> 4)) << 4) + (kk & 15)] = v;
      }
    }
    __syncthreads();
    v16bf af = frag_ld(As + (wid << 9));
#pragma unroll
    for (int ct = 0; ct < 4; ++ct) {
      v16bf bfg = frag_ld(Bs + (ct << 9));
      acc[ct] = wmma_bf16(af, bfg, acc[ct]);
    }
    __syncthreads();
  }

  int lane = tid & 31;
  int n = lane & 15;
  int ra = (lane >> 4) * 8;   // C layout: vgpr r -> M = r + 8*(lane>=16), N = lane&15
#pragma unroll
  for (int ct = 0; ct < 4; ++ct) {
#pragma unroll
    for (int r = 0; r < 8; ++r) {
      int row = m0 + wid * 16 + ra + r;
      int col = n0 + ct * 16 + n;
      if (row < M && col < N) {
        float v = acc[ct][r] * scale;
        if (bias) v += bias[col];
        if (relu) v = fmaxf(v, 0.f);
        long long o = coff + (long long)row * ldc + col;
        if (Cf) { if (accum) Cf[o] += v; else Cf[o] = v; }
        if (Cb) Cb[o] = (bf16_t)v;
      }
    }
  }
}

// ---------------- Small utility kernels ---------------------------------------

__global__ void k_f2b(const float* __restrict__ s, bf16_t* __restrict__ d, long long n) {
  long long i = (long long)blockIdx.x * 256 + threadIdx.x;
  if (i < n) d[i] = (bf16_t)s[i];
}
__global__ void k_zero_b(bf16_t* d, long long n) {
  long long i = (long long)blockIdx.x * 256 + threadIdx.x;
  if (i < n) d[i] = (bf16_t)0.f;
}
__global__ void k_copy_f(const float* __restrict__ s, float* __restrict__ d, int n) {
  int i = blockIdx.x * 256 + threadIdx.x;
  if (i < n) d[i] = s[i];
}

// layernorm over rows of 512 f32 -> bf16 (rows written at row_off offset)
__global__ __launch_bounds__(256)
void k_layernorm(const float* __restrict__ x, const float* __restrict__ w,
                 const float* __restrict__ b, bf16_t* __restrict__ out, int row_off)
{
  int r = blockIdx.x;
  const float* xr = x + (long long)r * DIMC;
  __shared__ float red[256];
  int t = threadIdx.x;
  float v0 = xr[t], v1 = xr[t + 256];
  red[t] = v0 + v1;
  __syncthreads();
  for (int s = 128; s > 0; s >>= 1) { if (t < s) red[t] += red[t + s]; __syncthreads(); }
  float mu = red[0] * (1.f / DIMC);
  __syncthreads();
  float d0 = v0 - mu, d1 = v1 - mu;
  red[t] = d0 * d0 + d1 * d1;
  __syncthreads();
  for (int s = 128; s > 0; s >>= 1) { if (t < s) red[t] += red[t + s]; __syncthreads(); }
  float rstd = rsqrtf(red[0] * (1.f / DIMC) + 1e-5f);
  bf16_t* o = out + (long long)(r + row_off) * DIMC;
  o[t]       = (bf16_t)(d0 * rstd * w[t] + b[t]);
  o[t + 256] = (bf16_t)(d1 * rstd * w[t + 256] + b[t + 256]);
}

// landmark means: dst[h][m][d] = mean_j src[(m*157+j)*512 + h*64 + d]
__global__ void k_landmark(const bf16_t* __restrict__ src, float* __restrict__ dst) {
  int idx = blockIdx.x * 256 + threadIdx.x;     // 8*256*64
  if (idx >= NHEAD * MLM * DHEAD) return;
  int d = idx & 63, m = (idx >> 6) & 255, h = idx >> 14;
  float s = 0.f;
  for (int j = 0; j < LVAL; ++j)
    s += (float)src[(long long)(m * LVAL + j) * DIMC + h * DHEAD + d];
  dst[idx] = s * (1.f / (float)LVAL);
}

// klT[h][d][m] = (bf16) kl[h][m][d]
__global__ void k_kltb(const float* __restrict__ kl, bf16_t* __restrict__ klT) {
  int idx = blockIdx.x * 256 + threadIdx.x;     // 8*64*256
  if (idx >= NHEAD * DHEAD * MLM) return;
  int m = idx & 255, d = (idx >> 8) & 63, h = idx >> 14;
  klT[idx] = (bf16_t)kl[(h << 14) + m * DHEAD + d];
}

// attn2[h][m][:] = softmax(q_l[h][m] . k_l[h][c]) over c  (f32 + bf16 copy)
__global__ __launch_bounds__(64)
void k_attn2(const float* __restrict__ ql, const float* __restrict__ kl,
             float* __restrict__ a2, bf16_t* __restrict__ a2b)
{
  int hm = blockIdx.x; int h = hm >> 8; int m = hm & 255;
  __shared__ float qsh[DHEAD];
  __shared__ float sl[MLM];
  __shared__ float red[64];
  int t = threadIdx.x;
  if (t < DHEAD) qsh[t] = ql[((h << 8) + m) * DHEAD + t];
  __syncthreads();
  for (int c = t; c < MLM; c += 64) {
    const float* kr = kl + ((h << 8) + c) * DHEAD;
    float s = 0.f;
#pragma unroll
    for (int d = 0; d < DHEAD; ++d) s += qsh[d] * kr[d];
    sl[c] = s;
  }
  __syncthreads();
  float lm = -1e30f;
  for (int c = t; c < MLM; c += 64) lm = fmaxf(lm, sl[c]);
  red[t] = lm; __syncthreads();
  for (int s = 32; s > 0; s >>= 1) { if (t < s) red[t] = fmaxf(red[t], red[t + s]); __syncthreads(); }
  float mx = red[0]; __syncthreads();
  float ls = 0.f;
  for (int c = t; c < MLM; c += 64) ls += __expf(sl[c] - mx);
  red[t] = ls; __syncthreads();
  for (int s = 32; s > 0; s >>= 1) { if (t < s) red[t] += red[t + s]; __syncthreads(); }
  float inv = 1.f / red[0];
  for (int c = t; c < MLM; c += 64) {
    float p = __expf(sl[c] - mx) * inv;
    long long o = (((long long)(h << 8) + m) << 8) + c;
    a2[o] = p; a2b[o] = (bf16_t)p;
  }
}

// abs row/col sums of attn2 per head
__global__ __launch_bounds__(256)
void k_rowcol(const float* __restrict__ a2, float* __restrict__ rs, float* __restrict__ cs) {
  int h = blockIdx.x, t = threadIdx.x;
  const float* Ah = a2 + (long long)h * MLM * MLM;
  float r = 0.f, c = 0.f;
  for (int j = 0; j < MLM; ++j) { r += fabsf(Ah[t * MLM + j]); c += fabsf(Ah[j * MLM + t]); }
  rs[h * MLM + t] = r; cs[h * MLM + t] = c;
}

// global (over heads) maxima -> scal[0] = 1/(rmax*cmax)
__global__ __launch_bounds__(256)
void k_scal(const float* __restrict__ rs, const float* __restrict__ cs, float* __restrict__ scal) {
  __shared__ float r1[256], r2[256];
  int t = threadIdx.x;
  float a = -1e30f, b = -1e30f;
  for (int i = t; i < NHEAD * MLM; i += 256) { a = fmaxf(a, rs[i]); b = fmaxf(b, cs[i]); }
  r1[t] = a; r2[t] = b; __syncthreads();
  for (int s = 128; s > 0; s >>= 1) {
    if (t < s) { r1[t] = fmaxf(r1[t], r1[t + s]); r2[t] = fmaxf(r2[t], r2[t + s]); }
    __syncthreads();
  }
  if (t == 0) scal[0] = 1.f / (r1[0] * r2[0]);
}

// z0 = attn2^T * scal (bf16)
__global__ void k_zinit(const float* __restrict__ a2, const float* __restrict__ scal,
                        bf16_t* __restrict__ zb) {
  int idx = blockIdx.x * 256 + threadIdx.x;       // 8*256*256
  if (idx >= NHEAD * MLM * MLM) return;
  int h = idx >> 16, i = (idx >> 8) & 255, j = idx & 255;
  float v = a2[((long long)h << 16) + (j << 8) + i] * scal[0];
  zb[idx] = (bf16_t)v;
}

// out = c*I - in  (bf16)
__global__ void k_pinv_elem(const float* __restrict__ inf, float c, bf16_t* __restrict__ outb) {
  int idx = blockIdx.x * 256 + threadIdx.x;
  if (idx >= NHEAD * MLM * MLM) return;
  int i = (idx >> 8) & 255, j = idx & 255;
  outb[idx] = (bf16_t)(c * ((i == j) ? 1.f : 0.f) - inf[idx]);
}

// attn3 @ v : per (head, landmark) streaming softmax over N (logits in 160KB LDS)
__global__ __launch_bounds__(256)
void k_attn3v(const float* __restrict__ ql, const bf16_t* __restrict__ kb,
              const bf16_t* __restrict__ vb, float* __restrict__ a3v,
              bf16_t* __restrict__ a3vb, int N)
{
  extern __shared__ float sm[];
  float* qsh = sm;            // 64
  float* red = sm + 64;       // 256
  float* lg  = sm + 320;      // N floats
  int hm = blockIdx.x; int h = hm >> 8; int m = hm & 255;
  int t = threadIdx.x;
  if (t < DHEAD) qsh[t] = ql[((h << 8) + m) * DHEAD + t];
  __syncthreads();
  float lmax = -1e30f;
  for (int j = t; j < N; j += 256) {
    const bf16_t* kr = kb + (long long)j * DIMC + h * DHEAD;
    float s = 0.f;
#pragma unroll
    for (int d = 0; d < DHEAD; ++d) s += qsh[d] * (float)kr[d];
    lg[j] = s; lmax = fmaxf(lmax, s);
  }
  red[t] = lmax; __syncthreads();
  for (int s = 128; s > 0; s >>= 1) { if (t < s) red[t] = fmaxf(red[t], red[t + s]); __syncthreads(); }
  float mx = red[0]; __syncthreads();
  float lsum = 0.f;
  for (int j = t; j < N; j += 256) { float e = __expf(lg[j] - mx); lg[j] = e; lsum += e; }
  red[t] = lsum; __syncthreads();
  for (int s = 128; s > 0; s >>= 1) { if (t < s) red[t] += red[t + s]; __syncthreads(); }
  float inv = 1.f / red[0];
  float acc[DHEAD];
#pragma unroll
  for (int d = 0; d < DHEAD; ++d) acc[d] = 0.f;
  for (int j = t; j < N; j += 256) {
    float p = lg[j];
    const bf16_t* vr = vb + (long long)j * DIMC + h * DHEAD;
#pragma unroll
    for (int d = 0; d < DHEAD; ++d) acc[d] += p * (float)vr[d];
  }
  __syncthreads();
  float* red2 = lg;           // reuse logits region as 256x64 reduction space
#pragma unroll
  for (int d = 0; d < DHEAD; ++d) red2[t * DHEAD + d] = acc[d];
  __syncthreads();
  for (int s = 128; s > 0; s >>= 1) {
    if (t < s) {
#pragma unroll
      for (int d = 0; d < DHEAD; ++d) red2[t * DHEAD + d] += red2[(t + s) * DHEAD + d];
    }
    __syncthreads();
  }
  if (t < DHEAD) {
    float o = red2[t] * inv;
    int oi = ((h << 8) + m) * DHEAD + t;
    a3v[oi] = o; a3vb[oi] = (bf16_t)o;
  }
}

// fused: out[128 rows, head] = softmax(q @ k_l^T) @ W2
// 8 waves per block; klT/W2 staged fragment-ready once per block; softmax done
// in accumulator registers with wave32 half-shuffles; P written directly in
// A-fragment layout (no f32 S round-trip).
__global__ __launch_bounds__(256)
void k_attn1(const bf16_t* __restrict__ qb, const bf16_t* __restrict__ klT,
             const bf16_t* __restrict__ W2b, float* __restrict__ ao)
{
  extern __shared__ char smem_raw[];
  bf16_t* Bk  = (bf16_t*)smem_raw;           // 16384 : klT fragments [ct*2+ks][lane][e]
  bf16_t* B2s = Bk + 16384;                  // 16384 : W2 fragments  [ct*8+ks][lane][e]
  bf16_t* qs  = B2s + 16384;                 //  8192 : q fragments   [wid*2+ks][lane][e]
  bf16_t* Ps  = qs + 8192;                   // 32768 : P fragments   [wid*8+ks][lane][e]

  int i0 = blockIdx.x * 128;
  int h  = blockIdx.y;
  int tid = threadIdx.x;
  int wid = tid >> 5;
  int lane = tid & 31;

  // stage klT (64x256, row=K) -> B fragments (paired along N)
  const bf16_t* klTh = klT + (long long)h * DHEAD * MLM;
#pragma unroll
  for (int it = 0; it < 32; ++it) {
    int g = tid + it * 256;        // pair index 0..8191
    int d = g >> 7;                // 0..63
    int mp = (g & 127) << 1;       // 0,2..254
    const bf16_t* src = klTh + d * MLM + mp;
    bf16_t v0 = src[0], v1 = src[1];
    int ks = d >> 5, kk = d & 31;
    int ct = mp >> 4, nn = mp & 15;
    int base = ((ct * 2 + ks) << 9) + (kk & 15) + ((16 * (kk >> 4)) << 4);
    Bk[base + (nn << 4)] = v0;
    Bk[base + ((nn + 1) << 4)] = v1;
  }
  // stage W2 (256x64, row=K) -> B fragments (paired along N)
  const bf16_t* w2h = W2b + (long long)h * MLM * DHEAD;
#pragma unroll
  for (int it = 0; it < 32; ++it) {
    int g = tid + it * 256;        // pair index 0..8191
    int k = g >> 5;                // 0..255
    int np = (g & 31) << 1;        // 0,2..62
    const bf16_t* src = w2h + k * DHEAD + np;
    bf16_t v0 = src[0], v1 = src[1];
    int ks = k >> 5, kk = k & 31;
    int ct = np >> 4, nn = np & 15;
    int base = ((ct * 8 + ks) << 9) + (kk & 15) + ((16 * (kk >> 4)) << 4);
    B2s[base + (nn << 4)] = v0;
    B2s[base + ((nn + 1) << 4)] = v1;
  }
  // stage q (128x64) -> A fragments, aligned 16B chunks on both sides
#pragma unroll
  for (int it = 0; it < 4; ++it) {
    int g = tid + it * 256;        // chunk index 0..1023
    int rg = g >> 3;               // 0..127
    int c8 = (g & 7) << 3;         // 0,8..56
    const bf16_t* src = qb + (long long)(i0 + rg) * DIMC + h * DHEAD + c8;
    int w2 = rg >> 4, r = rg & 15;
    int ks = c8 >> 5, kk = c8 & 31;
    int pl = r + 16 * (kk >> 4);
    int e = kk & 15;
    *(uint4*)&qs[((w2 * 2 + ks) << 9) + (pl << 4) + e] = *(const uint4*)src;
  }
  __syncthreads();

  // S = q @ klT  (16 x 256 per wave)
  v8f sacc[16];
#pragma unroll
  for (int ct = 0; ct < 16; ++ct)
#pragma unroll
    for (int e = 0; e < 8; ++e) sacc[ct][e] = 0.f;
  v16bf a0 = frag_ld(qs + ((wid * 2 + 0) << 9));
  v16bf a1 = frag_ld(qs + ((wid * 2 + 1) << 9));
#pragma unroll
  for (int ct = 0; ct < 16; ++ct) {
    sacc[ct] = wmma_bf16(a0, frag_ld(Bk + ((ct * 2 + 0) << 9)), sacc[ct]);
    sacc[ct] = wmma_bf16(a1, frag_ld(Bk + ((ct * 2 + 1) << 9)), sacc[ct]);
  }

  // register softmax: row m = r + 8*(lane>=16); cols spread over ct (in-lane)
  // and n = lane&15 (cross-lane within each 16-lane half)
  float mx[8], inv[8];
#pragma unroll
  for (int r = 0; r < 8; ++r) {
    float m1 = -1e30f;
#pragma unroll
    for (int ct = 0; ct < 16; ++ct) m1 = fmaxf(m1, sacc[ct][r]);
#pragma unroll
    for (int d = 1; d < 16; d <<= 1) m1 = fmaxf(m1, __shfl_xor(m1, d, 32));
    float s1 = 0.f;
#pragma unroll
    for (int ct = 0; ct < 16; ++ct) s1 += __expf(sacc[ct][r] - m1);
#pragma unroll
    for (int d = 1; d < 16; d <<= 1) s1 += __shfl_xor(s1, d, 32);
    mx[r] = m1; inv[r] = 1.f / s1;
  }
  // store P directly in A-fragment layout
  {
    int n = lane & 15, hi = lane >> 4;
#pragma unroll
    for (int ct = 0; ct < 16; ++ct) {
      int k = ct * 16 + n;
      int ks = k >> 5, kk = k & 31;
      int e = kk & 15;
      int lhi = 16 * (kk >> 4);
#pragma unroll
      for (int r = 0; r < 8; ++r) {
        float p = __expf(sacc[ct][r] - mx[r]) * inv[r];
        int m = r + 8 * hi;
        Ps[((wid * 8 + ks) << 9) + ((m + lhi) << 4) + e] = (bf16_t)p;
      }
    }
  }
  __syncthreads();

  // out = P @ W2  (16 x 64 per wave)
  int n = lane & 15, ra = (lane >> 4) * 8;
#pragma unroll
  for (int ct2 = 0; ct2 < 4; ++ct2) {
    v8f acc;
#pragma unroll
    for (int e = 0; e < 8; ++e) acc[e] = 0.f;
#pragma unroll
    for (int ks = 0; ks < 8; ++ks) {
      v16bf a = frag_ld(Ps + ((wid * 8 + ks) << 9));
      v16bf b = frag_ld(B2s + ((ct2 * 8 + ks) << 9));
      acc = wmma_bf16(a, b, acc);
    }
#pragma unroll
    for (int r = 0; r < 8; ++r)
      ao[(long long)(i0 + wid * 16 + ra + r) * DIMC + h * DHEAD + ct2 * 16 + n] = acc[r];
  }
}

// depthwise 33x1 conv over sequence per head, accumulate into attn output
__global__ void k_dwres(const bf16_t* __restrict__ vb, const float* __restrict__ w,
                        float* __restrict__ ao, int N)
{
  long long idx = (long long)blockIdx.x * 256 + threadIdx.x;
  if (idx >= (long long)N * DIMC) return;
  int c = (int)(idx & 511);
  long long nrow = idx >> 9;
  int h = c >> 6;
  float s = 0.f;
  for (int t = 0; t < 33; ++t) {
    long long j = nrow + t - 16;
    if (j >= 0 && j < N) s += w[h * 33 + t] * (float)vb[j * DIMC + c];
  }
  ao[idx] += s;
}

// PPEG: h2[1+p] = f + conv7+b7 + conv5+b5 + conv3+b3 on 200x200 grid
__global__ void k_ppeg(const float* __restrict__ hin, float* __restrict__ hout,
                       const float* __restrict__ w7, const float* __restrict__ b7,
                       const float* __restrict__ w5, const float* __restrict__ b5,
                       const float* __restrict__ w3, const float* __restrict__ b3)
{
  long long idx = (long long)blockIdx.x * 256 + threadIdx.x;
  if (idx >= (long long)NPATCH * DIMC) return;
  int c = (int)(idx & 511);
  long long p = idx >> 9;
  int y = (int)(p / 200), x = (int)(p % 200);
  float v = hin[(1 + p) * DIMC + c] + b7[c] + b5[c] + b3[c];
  for (int ky = 0; ky < 7; ++ky)
    for (int kx = 0; kx < 7; ++kx) {
      int yy = y + ky - 3, xx = x + kx - 3;
      if (yy >= 0 && yy < 200 && xx >= 0 && xx < 200)
        v += w7[c * 49 + ky * 7 + kx] * hin[(long long)(1 + yy * 200 + xx) * DIMC + c];
    }
  for (int ky = 0; ky < 5; ++ky)
    for (int kx = 0; kx < 5; ++kx) {
      int yy = y + ky - 2, xx = x + kx - 2;
      if (yy >= 0 && yy < 200 && xx >= 0 && xx < 200)
        v += w5[c * 25 + ky * 5 + kx] * hin[(long long)(1 + yy * 200 + xx) * DIMC + c];
    }
  for (int ky = 0; ky < 3; ++ky)
    for (int kx = 0; kx < 3; ++kx) {
      int yy = y + ky - 1, xx = x + kx - 1;
      if (yy >= 0 && yy < 200 && xx >= 0 && xx < 200)
        v += w3[c * 9 + ky * 3 + kx] * hin[(long long)(1 + yy * 200 + xx) * DIMC + c];
    }
  hout[(1 + p) * DIMC + c] = v;
}

// final: layernorm(cls row) @ fc2 + fc2_b
__global__ __launch_bounds__(512)
void k_head(const float* __restrict__ h, const float* __restrict__ nw,
            const float* __restrict__ nb, const float* __restrict__ fw,
            const float* __restrict__ fb, float* __restrict__ out)
{
  __shared__ float red[512];
  int t = threadIdx.x;
  float v = h[t];
  red[t] = v; __syncthreads();
  for (int s = 256; s > 0; s >>= 1) { if (t < s) red[t] += red[t + s]; __syncthreads(); }
  float mu = red[0] * (1.f / 512.f); __syncthreads();
  float d = v - mu;
  red[t] = d * d; __syncthreads();
  for (int s = 256; s > 0; s >>= 1) { if (t < s) red[t] += red[t + s]; __syncthreads(); }
  float rstd = rsqrtf(red[0] * (1.f / 512.f) + 1e-5f); __syncthreads();
  float xn = d * rstd * nw[t] + nb[t];
  red[t] = xn * fw[t * 2 + 0]; __syncthreads();
  for (int s = 256; s > 0; s >>= 1) { if (t < s) red[t] += red[t + s]; __syncthreads(); }
  if (t == 0) out[0] = red[0] + fb[0];
  __syncthreads();
  red[t] = xn * fw[t * 2 + 1]; __syncthreads();
  for (int s = 256; s > 0; s >>= 1) { if (t < s) red[t] += red[t + s]; __syncthreads(); }
  if (t == 0) out[1] = red[0] + fb[1];
}

// ---------------- Host-side orchestration ------------------------------------

extern "C" void kernel_launch(void* const* d_in, const int* in_sizes, int n_in,
                              void* d_out, int out_size, void* d_ws, size_t ws_size,
                              hipStream_t stream)
{
  (void)in_sizes; (void)n_in; (void)out_size; (void)ws_size;
  const float* x      = (const float*)d_in[0];
  const float* fc1_w  = (const float*)d_in[1];
  const float* fc1_b  = (const float*)d_in[2];
  const float* cls    = (const float*)d_in[3];
  const float* l1_nw  = (const float*)d_in[4];
  const float* l1_nb  = (const float*)d_in[5];
  const float* l1_qkv = (const float*)d_in[6];
  const float* l1_ow  = (const float*)d_in[7];
  const float* l1_ob  = (const float*)d_in[8];
  const float* l1_rw  = (const float*)d_in[9];
  const float* p7w = (const float*)d_in[10]; const float* p7b = (const float*)d_in[11];
  const float* p5w = (const float*)d_in[12]; const float* p5b = (const float*)d_in[13];
  const float* p3w = (const float*)d_in[14]; const float* p3b = (const float*)d_in[15];
  const float* l2_nw  = (const float*)d_in[16];
  const float* l2_nb  = (const float*)d_in[17];
  const float* l2_qkv = (const float*)d_in[18];
  const float* l2_ow  = (const float*)d_in[19];
  const float* l2_ob  = (const float*)d_in[20];
  const float* l2_rw  = (const float*)d_in[21];
  const float* nw     = (const float*)d_in[22];
  const float* nb     = (const float*)d_in[23];
  const float* fc2w   = (const float*)d_in[24];
  const float* fc2b   = (const float*)d_in[25];
  float* out = (float*)d_out;

  char* ws = (char*)d_ws;
  size_t off = 0;
  auto alloc = [&](size_t bytes) { size_t o = off; off += (bytes + 255) & ~(size_t)255; return o; };

  float*  H    = (float*)(ws + alloc((size_t)NTOK * DIMC * 4));
  float*  H2   = (float*)(ws + alloc((size_t)NTOK * DIMC * 4));
  bf16_t* XLN  = (bf16_t*)(ws + alloc((size_t)NPAD * DIMC * 2));   // alias: attn-out bf16
  bf16_t* QB   = (bf16_t*)(ws + alloc((size_t)NPAD * DIMC * 2));
  bf16_t* KB   = (bf16_t*)(ws + alloc((size_t)NPAD * DIMC * 2));
  bf16_t* VB   = (bf16_t*)(ws + alloc((size_t)NPAD * DIMC * 2));
  float*  AO   = (float*)(ws + alloc((size_t)NPAD * DIMC * 4));    // alias: x bf16
  bf16_t* WQKVB= (bf16_t*)(ws + alloc((size_t)DIMC * 3 * DIMC * 2)); // alias: fc1w bf16
  bf16_t* WOUTB= (bf16_t*)(ws + alloc((size_t)DIMC * DIMC * 2));
  float*  QL   = (float*)(ws + alloc((size_t)NHEAD * MLM * DHEAD * 4));
  float*  KL   = (float*)(ws + alloc((size_t)NHEAD * MLM * DHEAD * 4));
  bf16_t* KLTB = (bf16_t*)(ws + alloc((size_t)NHEAD * DHEAD * MLM * 2));
  float*  A2   = (float*)(ws + alloc((size_t)NHEAD * MLM * MLM * 4));
  bf16_t* XBH  = (bf16_t*)(ws + alloc((size_t)NHEAD * MLM * MLM * 2));
  bf16_t* ZB   = (bf16_t*)(ws + alloc((size_t)NHEAD * MLM * MLM * 2));
  bf16_t* ZB2  = (bf16_t*)(ws + alloc((size_t)NHEAD * MLM * MLM * 2));
  bf16_t* T1B  = (bf16_t*)(ws + alloc((size_t)NHEAD * MLM * MLM * 2));
  bf16_t* T3B  = (bf16_t*)(ws + alloc((size_t)NHEAD * MLM * MLM * 2));
  float*  T2F  = (float*)(ws + alloc((size_t)NHEAD * MLM * MLM * 4));
  float*  RS   = (float*)(ws + alloc((size_t)NHEAD * MLM * 4));
  float*  CS   = (float*)(ws + alloc((size_t)NHEAD * MLM * 4));
  float*  SCAL = (float*)(ws + alloc(256));
  float*  A3V  = (float*)(ws + alloc((size_t)NHEAD * MLM * DHEAD * 4));
  bf16_t* A3VB = (bf16_t*)(ws + alloc((size_t)NHEAD * MLM * DHEAD * 2));
  bf16_t* W2B  = (bf16_t*)(ws + alloc((size_t)NHEAD * MLM * DHEAD * 2));
  bf16_t* XB   = (bf16_t*)AO;   // x bf16 lives where AO will later live
  bf16_t* AOB  = XLN;           // attn-out bf16 reuses xln region

  auto gemm = [&](const bf16_t* A, int lda, long long sA,
                  const bf16_t* B, int ldb, long long sB,
                  float* Cf, bf16_t* Cb, int ldc, long long sC,
                  int M, int N, int K, const float* bias,
                  int relu, int accum, float scale, int batch) {
    dim3 g((N + BNt - 1) / BNt, (M + BMt - 1) / BMt, batch);
    k_gemm<<<g, 256, 0, stream>>>(A, lda, sA, B, ldb, sB, Cf, Cb, ldc, sC,
                                  M, N, K, bias, relu, accum, scale);
  };

  // ---- Stage 0: fc1 + relu + cls concat ----
  { long long n = (long long)NPATCH * INDIM;
    k_f2b<<<(int)((n + 255) / 256), 256, 0, stream>>>(x, XB, n); }
  { long long n = (long long)INDIM * DIMC;
    k_f2b<<<(int)((n + 255) / 256), 256, 0, stream>>>(fc1_w, WQKVB, n); }
  k_copy_f<<<2, 256, 0, stream>>>(cls, H, DIMC);
  gemm(XB, INDIM, 0, WQKVB, DIMC, 0, H + DIMC, nullptr, DIMC, 0,
       NPATCH, DIMC, INDIM, fc1_b, 1, 0, 1.f, 1);

  // ---- One Nystrom attention layer (residual accumulated into h) ----
  auto layer = [&](float* h, const float* nw_, const float* nb_, const float* qkv_w,
                   const float* out_w, const float* out_b, const float* res_w) {
    { long long n = (long long)PADR * DIMC;
      k_zero_b<<<(int)((n + 255) / 256), 256, 0, stream>>>(XLN, n); }
    k_layernorm<<<NTOK, 256, 0, stream>>>(h, nw_, nb_, XLN, PADR);
    { long long n = (long long)DIMC * 3 * DIMC;
      k_f2b<<<(int)((n + 255) / 256), 256, 0, stream>>>(qkv_w, WQKVB, n); }
    { long long n = (long long)DIMC * DIMC;
      k_f2b<<<(int)((n + 255) / 256), 256, 0, stream>>>(out_w, WOUTB, n); }
    // q (scaled by dh^-0.5), k, v
    gemm(XLN, DIMC, 0, WQKVB + 0 * DIMC, 3 * DIMC, 0, nullptr, QB, DIMC, 0,
         NPAD, DIMC, DIMC, nullptr, 0, 0, 0.125f, 1);
    gemm(XLN, DIMC, 0, WQKVB + 1 * DIMC, 3 * DIMC, 0, nullptr, KB, DIMC, 0,
         NPAD, DIMC, DIMC, nullptr, 0, 0, 1.f, 1);
    gemm(XLN, DIMC, 0, WQKVB + 2 * DIMC, 3 * DIMC, 0, nullptr, VB, DIMC, 0,
         NPAD, DIMC, DIMC, nullptr, 0, 0, 1.f, 1);
    k_landmark<<<512, 256, 0, stream>>>(QB, QL);
    k_landmark<<<512, 256, 0, stream>>>(KB, KL);
    k_kltb<<<512, 256, 0, stream>>>(KL, KLTB);
    k_attn2<<<NHEAD * MLM, 64, 0, stream>>>(QL, KL, A2, XBH);
    k_rowcol<<<NHEAD, 256, 0, stream>>>(A2, RS, CS);
    k_scal<<<1, 256, 0, stream>>>(RS, CS, SCAL);
    k_zinit<<<2048, 256, 0, stream>>>(A2, SCAL, ZB);
    // attn3 @ v (streaming softmax with logits in LDS)
    {
      size_t smem = (size_t)(320 + NPAD) * 4;
      hipFuncSetAttribute(reinterpret_cast<const void*>(k_attn3v),
                          hipFuncAttributeMaxDynamicSharedMemorySize, (int)smem);
      k_attn3v<<<NHEAD * MLM, 256, smem, stream>>>(QL, KB, VB, A3V, A3VB, NPAD);
    }
    // Moore-Penrose pinv: 6 iterations, batched 256^3 WMMA GEMMs
    bf16_t* zin = ZB; bf16_t* zout = ZB2;
    const long long SQ = (long long)MLM * MLM;
    for (int it = 0; it < 6; ++it) {
      gemm(XBH, MLM, SQ, zin, MLM, SQ, T2F, T1B, MLM, SQ, MLM, MLM, MLM,
           nullptr, 0, 0, 1.f, NHEAD);                                   // xz
      k_pinv_elem<<<2048, 256, 0, stream>>>(T2F, 7.f, T3B);              // 7I - xz
      gemm(T1B, MLM, SQ, T3B, MLM, SQ, T2F, nullptr, MLM, SQ, MLM, MLM, MLM,
           nullptr, 0, 0, 1.f, NHEAD);                                   // xz @ t
      k_pinv_elem<<<2048, 256, 0, stream>>>(T2F, 15.f, T3B);             // 15I - ..
      gemm(T1B, MLM, SQ, T3B, MLM, SQ, T2F, nullptr, MLM, SQ, MLM, MLM, MLM,
           nullptr, 0, 0, 1.f, NHEAD);                                   // xz @ w
      k_pinv_elem<<<2048, 256, 0, stream>>>(T2F, 13.f, T3B);             // 13I - ..
      gemm(zin, MLM, SQ, T3B, MLM, SQ, nullptr, zout, MLM, SQ, MLM, MLM, MLM,
           nullptr, 0, 0, 0.25f, NHEAD);                                 // z'
      bf16_t* tmp = zin; zin = zout; zout = tmp;
    }
    // W2 = pinv @ (attn3 @ v)
    gemm(zin, MLM, SQ, A3VB, DHEAD, (long long)MLM * DHEAD,
         nullptr, W2B, DHEAD, (long long)MLM * DHEAD,
         MLM, DHEAD, MLM, nullptr, 0, 0, 1.f, NHEAD);
    // out = softmax(q @ k_l^T) @ W2  (fused, WMMA x2, fragment-ready LDS)
    {
      size_t smem = (size_t)(16384 + 16384 + 8192 + 32768) * 2;   // 147456 B
      hipFuncSetAttribute(reinterpret_cast<const void*>(k_attn1),
                          hipFuncAttributeMaxDynamicSharedMemorySize, (int)smem);
      k_attn1<<<dim3(NPAD / 128, NHEAD), 256, smem, stream>>>(QB, KLTB, W2B, AO);
    }
    // + depthwise 33-tap residual conv on v
    { long long n = (long long)NPAD * DIMC;
      k_dwres<<<(int)((n + 255) / 256), 256, 0, stream>>>(VB, res_w, AO, NPAD); }
    { long long n = (long long)NPAD * DIMC;
      k_f2b<<<(int)((n + 255) / 256), 256, 0, stream>>>(AO, AOB, n); }
    // h += (attn[-NTOK:] @ out_w + out_b)
    gemm(AOB + (long long)PADR * DIMC, DIMC, 0, WOUTB, DIMC, 0, h, nullptr, DIMC, 0,
         NTOK, DIMC, DIMC, out_b, 0, 1, 1.f, 1);
  };

  layer(H, l1_nw, l1_nb, l1_qkv, l1_ow, l1_ob, l1_rw);

  { long long n = (long long)NPATCH * DIMC;
    k_ppeg<<<(int)((n + 255) / 256), 256, 0, stream>>>(H, H2, p7w, p7b, p5w, p5b, p3w, p3b); }
  k_copy_f<<<2, 256, 0, stream>>>(H, H2, DIMC);   // cls token passthrough

  layer(H2, l2_nw, l2_nb, l2_qkv, l2_ow, l2_ob, l2_rw);

  k_head<<<1, 512, 0, stream>>>(H2, nw, nb, fc2w, fc2b, out);
}
